// HypergraphConstruction_44925357916947
// MI455X (gfx1250) — matrix operand order repaired and checked by reading, not compiled, not run on hardware
//
#include <hip/hip_runtime.h>

// ---------------- CDNA5 WMMA plumbing (wave32, 16x16x32 bf16) ----------------
typedef __attribute__((ext_vector_type(16))) __bf16 v16bf;
typedef __attribute__((ext_vector_type(8)))  __bf16 v8bf;
typedef __attribute__((ext_vector_type(8)))  float  v8f;

__device__ __forceinline__ v8f wmma_bf16(v16bf a, v16bf b, v8f c) {
  // 8 args: (neg_a, A, neg_b, B, c_mod, C, reuse_a, reuse_b)
  return __builtin_amdgcn_wmma_f32_16x16x32_bf16(false, a, false, b, (short)0, c, false, false);
}

// A-matrix 16x32 bf16 fragment (ISA 7.12.2): lane<16 -> row=lane, K={0..7,16..23};
// lane>=16 -> row=lane-16, K={8..15,24..31}. Both halves contiguous -> 2x ds_load_b128.
__device__ __forceinline__ v16bf load_a16x32(const __bf16* A, int lda, int row0, int k0) {
  int lane = threadIdx.x & 31;
  const __bf16* rp = A + (row0 + (lane & 15)) * lda + k0 + ((lane & 16) ? 8 : 0);
  v16bf a;
#pragma unroll
  for (int j = 0; j < 8; ++j) { a[j] = rp[j]; a[8 + j] = rp[16 + j]; }
  return a;
}

// ---- packed (fragment-major) B storage ----
// B-matrix 32x16 tile: lane = ((k>>3)&1)*16 + (n&15); element within lane = ((k>>4)&1)*8 + (k&7).
// Each lane's 16 bf16 stored contiguously (32B chunk) -> fragment load = one v16bf read.
__device__ __forceinline__ int bpack(int k, int n, int ntiles) {
  return ((((k >> 5) * ntiles + (n >> 4)) * 2 + ((k >> 3) & 1)) * 16 + (n & 15)) * 16 + ((k >> 4) & 1) * 8 + (k & 7);
}
__device__ __forceinline__ void stage_pack(const float* __restrict__ src, __bf16* __restrict__ dst,
                                           int K, int N, int ntiles, int tid, int nthreads) {
  for (int i = tid; i < K * N; i += nthreads)
    dst[bpack(i / N, i % N, ntiles)] = (__bf16)src[i];
}
__device__ __forceinline__ v16bf load_bp(const __bf16* P, int tile) {
  int lane = threadIdx.x & 31;
  return *(const v16bf*)(P + (tile * 32 + lane) * 16);
}

#define BN_INV 0.99999499999f /* 1/sqrt(1+1e-5) */

// ---------------- problem dims ----------------
#define BM 32
#define PP 768
#define PT 20
#define CIN 9
#define HH 64
#define OUTF 256
#define NPOLY (BM * PP) /* 24576 */
#define BB 2048
#define TT 20
#define FF 8
#define LHID 128
#define G4 512
#define NE 32
#define DD 128

static __device__ const v8f VZ8 = {0.f, 0.f, 0.f, 0.f, 0.f, 0.f, 0.f, 0.f};

// =====================================================================
// Kernel 1: fused PolylineEncoder up to second max-pool (per polyline).
// =====================================================================
__global__ __launch_bounds__(128) void pe_encode_kernel(
    const float* __restrict__ xg, const unsigned char* __restrict__ maskg,
    const float* __restrict__ preW, const float* __restrict__ preG, const float* __restrict__ preB,
    const float* __restrict__ W1, const float* __restrict__ g1, const float* __restrict__ b1,
    const float* __restrict__ W2, const float* __restrict__ g2, const float* __restrict__ b2,
    float* __restrict__ pooled2, float* __restrict__ validf) {
  __shared__ float xs[PT][CIN];
  __shared__ float msk[PT];
  __shared__ alignas(32) __bf16 cat[32][128];   // [feat(64) | pooled bcast(64)]
  __shared__ alignas(32) __bf16 Wbuf[128 * 64]; // packed W1 (16 tiles) then W2 (8 tiles)
  __shared__ alignas(32) __bf16 h1s[32][64];
  __shared__ float pool1[64];
  __shared__ float pool2[64];
  __shared__ float anyv;

  const int poly = blockIdx.x;
  const int tid = threadIdx.x;
  const int wave = tid >> 5, lane = tid & 31;

  for (int i = tid; i < PT * CIN; i += 128) xs[i / CIN][i % CIN] = xg[poly * PT * CIN + i];
  for (int i = tid; i < PT; i += 128) msk[i] = maskg[poly * PT + i] ? 1.f : 0.f;
  for (int i = tid; i < 32 * 128; i += 128) (&cat[0][0])[i] = (__bf16)0.f;
  __syncthreads();

  if (tid == 0) {
    float any = 0.f;
    for (int p = 0; p < PT; ++p) any = fmaxf(any, msk[p]);
    anyv = any;
  }
  // ---- pre-MLP (K=9, VALU) + masked first max-pool ----
  if (tid < 64) {
    const int c = tid;
    float wc[CIN];
#pragma unroll
    for (int k = 0; k < CIN; ++k) wc[k] = preW[k * HH + c];
    const float ga = preG[c] * BN_INV, be = preB[c];
    float pmax = 0.f;
    for (int p = 0; p < PT; ++p) {
      float acc = 0.f;
#pragma unroll
      for (int k = 0; k < CIN; ++k) acc += xs[p][k] * wc[k];
      float v = fmaxf(acc * ga + be, 0.f) * msk[p];
      cat[p][c] = (__bf16)v;
      pmax = fmaxf(pmax, v);
    }
    pool1[c] = pmax;
  }
  __syncthreads();
  for (int i = tid; i < PT * 64; i += 128) cat[i >> 6][64 + (i & 63)] = (__bf16)pool1[i & 63];
  stage_pack(W1, Wbuf, 128, 64, 4, tid, 128);
  __syncthreads();

  // ---- mlp1: (32x128)@(128x64); wave w -> N-tile w ----
  {
    v8f a0 = VZ8, a1 = VZ8;
    for (int k0 = 0; k0 < 128; k0 += 32) {
      v16bf bf = load_bp(Wbuf, (k0 >> 5) * 4 + wave);
      a0 = wmma_bf16(load_a16x32(&cat[0][0], 128, 0, k0), bf, a0);
      a1 = wmma_bf16(load_a16x32(&cat[0][0], 128, 16, k0), bf, a1);
    }
    const int cn = wave * 16 + (lane & 15), mo = (lane & 16) ? 8 : 0;
    const float ga = g1[cn] * BN_INV, be = b1[cn];
#pragma unroll
    for (int v = 0; v < 8; ++v) {
      h1s[mo + v][cn]      = (__bf16)fmaxf(a0[v] * ga + be, 0.f);
      h1s[16 + mo + v][cn] = (__bf16)fmaxf(a1[v] * ga + be, 0.f);
    }
  }
  __syncthreads();
  stage_pack(W2, Wbuf, 64, 64, 4, tid, 128);
  __syncthreads();

  // ---- mlp2: (32x64)@(64x64) + masked second max-pool ----
  {
    v8f a0 = VZ8, a1 = VZ8;
    for (int k0 = 0; k0 < 64; k0 += 32) {
      v16bf bf = load_bp(Wbuf, (k0 >> 5) * 4 + wave);
      a0 = wmma_bf16(load_a16x32(&h1s[0][0], 64, 0, k0), bf, a0);
      a1 = wmma_bf16(load_a16x32(&h1s[0][0], 64, 16, k0), bf, a1);
    }
    const int cn = wave * 16 + (lane & 15), mo = (lane & 16) ? 8 : 0;
    const float ga = g2[cn] * BN_INV, be = b2[cn];
    float pmax = 0.f;
#pragma unroll
    for (int v = 0; v < 8; ++v) {
      pmax = fmaxf(pmax, fmaxf(a0[v] * ga + be, 0.f) * msk[mo + v]);
      const int r1 = 16 + mo + v;
      if (r1 < PT) pmax = fmaxf(pmax, fmaxf(a1[v] * ga + be, 0.f) * msk[r1]);
    }
    pmax = fmaxf(pmax, __shfl_xor(pmax, 16));
    if ((lane & 16) == 0) pool2[cn] = pmax;
  }
  __syncthreads();
  if (tid < 64) pooled2[poly * 64 + tid] = pool2[tid];
  if (tid == 64) validf[poly] = anyv;
}

// =====================================================================
// Kernel 2: polyline_features = (pooled2 @ pe_out_W + b) * valid
// =====================================================================
__global__ __launch_bounds__(128) void pe_out_kernel(
    const float* __restrict__ pooled2, const float* __restrict__ validf,
    const float* __restrict__ W, const float* __restrict__ bias, float* __restrict__ outp) {
  __shared__ alignas(32) __bf16 As[16][64];
  __shared__ alignas(32) __bf16 Bs[64 * 256]; // packed: 2 ktiles x 16 ntiles
  const int row0 = blockIdx.x * 16, tid = threadIdx.x;
  for (int i = tid; i < 16 * 64; i += 128)
    As[i >> 6][i & 63] = (__bf16)pooled2[(row0 + (i >> 6)) * 64 + (i & 63)];
  stage_pack(W, Bs, 64, 256, 16, tid, 128);
  __syncthreads();

  const int wave = tid >> 5, lane = tid & 31;
  v8f acc[4];
#pragma unroll
  for (int nt = 0; nt < 4; ++nt) acc[nt] = VZ8;
  for (int k0 = 0; k0 < 64; k0 += 32) {
    v16bf a = load_a16x32(&As[0][0], 64, 0, k0);
#pragma unroll
    for (int nt = 0; nt < 4; ++nt)
      acc[nt] = wmma_bf16(a, load_bp(Bs, (k0 >> 5) * 16 + wave * 4 + nt), acc[nt]);
  }
  const int cn = lane & 15, mo = (lane & 16) ? 8 : 0;
#pragma unroll
  for (int nt = 0; nt < 4; ++nt) {
    const int col = wave * 64 + nt * 16 + cn;
#pragma unroll
    for (int v = 0; v < 8; ++v) {
      const int row = row0 + mo + v;
      outp[row * 256 + col] = (acc[nt][v] + bias[col]) * validf[row];
    }
  }
}

// =====================================================================
// Kernel 3: LSTM. Wh packed bf16 (128KB) LDS-resident for all 20 steps
// (CDNA5 WGP has 320KB LDS). wave g computes gate g.
// =====================================================================
__global__ __launch_bounds__(128) void lstm_kernel(
    const float* __restrict__ x, const float* __restrict__ Wx, const float* __restrict__ Wh,
    const float* __restrict__ bias, float* __restrict__ hout) {
  extern __shared__ __align__(32) char smem[];
  __bf16* Wh_s = (__bf16*)smem;                                   // packed 128x512: 131072B
  __bf16* Wxp_s = (__bf16*)(smem + 131072);                       // packed 32x512 (K zero-pad): 32768B
  __bf16* xsb = (__bf16*)(smem + 131072 + 32768);                 // 16*20*8 bf16: 5120B
  __bf16* hs = (__bf16*)(smem + 131072 + 32768 + 5120);           // 16x128 bf16: 4096B
  float* cs = (float*)(smem + 131072 + 32768 + 5120 + 4096);      // 16x128 f32: 8192B
  float* gts = (float*)(smem + 131072 + 32768 + 5120 + 4096 + 8192); // 4x16x128 f32: 32768B

  const int type = blockIdx.y;
  const int row0 = blockIdx.x * 16;
  const int tid = threadIdx.x, wave = tid >> 5, lane = tid & 31;
  const float* Whp = Wh + type * LHID * G4;
  const float* Wxp = Wx + type * FF * G4;
  const float* bp = bias + type * G4;

  stage_pack(Whp, Wh_s, LHID, G4, 32, tid, 128);
  // pack Wx with K padded 8->32 (zeros), single pass: i = ((tile*32+lane)*16+elem)
  for (int i = tid; i < 32 * G4; i += 128) {
    const int e = i & 15, l = (i >> 4) & 31, tl = i >> 9;
    const int k = ((l >> 4) << 4) + e, n = tl * 16 + (l & 15);
    Wxp_s[i] = (k < FF) ? (__bf16)Wxp[k * G4 + n] : (__bf16)0.f;
  }
  for (int i = tid; i < 16 * TT * FF; i += 128) xsb[i] = (__bf16)x[row0 * TT * FF + i];
  for (int i = tid; i < 16 * LHID; i += 128) { hs[i] = (__bf16)0.f; cs[i] = 0.f; }
  __syncthreads();

  const int gbase = wave * 128;
  const int cn = lane & 15, mo = (lane & 16) ? 8 : 0;

  for (int t = 0; t < TT; ++t) {
    // A-fragment of x_t (16x8 zero-padded to 16x32); one 16B LDS read for lanes<16
    v16bf ax;
#pragma unroll
    for (int e = 0; e < 16; ++e) ax[e] = (__bf16)0.f;
    if (lane < 16) {
      const v8bf x8 = *(const v8bf*)&xsb[(lane * TT + t) * FF];
#pragma unroll
      for (int j = 0; j < 8; ++j) ax[j] = x8[j];
    }
    v8f acc[8];
#pragma unroll
    for (int nt = 0; nt < 8; ++nt)
      acc[nt] = wmma_bf16(ax, load_bp(Wxp_s, wave * 8 + nt), VZ8);  // xg = x_t @ Wx
    // + h @ Wh
    for (int k0 = 0; k0 < LHID; k0 += 32) {
      v16bf a = load_a16x32(hs, LHID, 0, k0);
#pragma unroll
      for (int nt = 0; nt < 8; ++nt)
        acc[nt] = wmma_bf16(a, load_bp(Wh_s, (k0 >> 5) * 32 + wave * 8 + nt), acc[nt]);
    }
    // bias + gate activation -> LDS
#pragma unroll
    for (int nt = 0; nt < 8; ++nt) {
      const int col = gbase + nt * 16 + cn;
#pragma unroll
      for (int v = 0; v < 8; ++v) {
        const int m = mo + v;
        float s = acc[nt][v] + bp[col];
        float a = (wave == 2) ? tanhf(s) : (1.f / (1.f + __expf(-s)));
        gts[(wave * 16 + m) * LHID + (nt * 16 + cn)] = a;
      }
    }
    __syncthreads();
    // c = f*c + i*g ; h = o*tanh(c)   (gate order i,f,g,o)
    for (int i = tid; i < 16 * LHID; i += 128) {
      float iv = gts[i], fv = gts[2048 + i], gv = gts[4096 + i], ov = gts[6144 + i];
      float c = fv * cs[i] + iv * gv;
      cs[i] = c;
      hs[i] = (__bf16)(ov * tanhf(c));
    }
    __syncthreads();
  }
  for (int i = tid; i < 16 * LHID; i += 128)
    hout[(BB * type + row0 + (i >> 7)) * LHID + (i & 127)] = (float)hs[i];
}

// =====================================================================
// Kernel 4: MCGBlock (3 blocks), one batch row per block.
// =====================================================================
__global__ __launch_bounds__(128) void mcg_kernel(
    const float* __restrict__ sdata, const float* __restrict__ sW,
    const float* __restrict__ sG, const float* __restrict__ sBt,
    const float* __restrict__ cW, const float* __restrict__ cG, const float* __restrict__ cBt,
    float* __restrict__ outp) {
  __shared__ float s32[NE * DD];
  __shared__ alignas(32) __bf16 sb[NE * DD];
  __shared__ alignas(32) __bf16 Wb[DD * DD]; // packed: 4 ktiles x 8 ntiles
  __shared__ float cvec[DD], cival[DD], aggc[DD];

  const int b = blockIdx.x, tid = threadIdx.x;
  const int wave = tid >> 5, lane = tid & 31;
  const int cn = lane & 15, mo = (lane & 16) ? 8 : 0;

  for (int i = tid; i < NE * DD; i += 128) {
    float v = sdata[b * NE * DD + i];
    s32[i] = v; sb[i] = (__bf16)v;
  }
  if (tid < DD) cvec[tid] = 1.f;
  __syncthreads();

  for (int blk = 0; blk < 3; ++blk) {
    if (tid < DD) {
      if (blk == 0) cival[tid] = cvec[tid];
      else {
        const float* wc = cW + (blk - 1) * DD * DD;
        float acc = 0.f;
        for (int k = 0; k < DD; ++k) acc += cvec[k] * wc[k * DD + tid];
        cival[tid] = fmaxf(acc * cG[(blk - 1) * DD + tid] * BN_INV + cBt[(blk - 1) * DD + tid], 0.f);
      }
      aggc[tid] = 0.f;
    }
    stage_pack(sW + blk * DD * DD, Wb, DD, DD, 8, tid, 128);
    __syncthreads();

    v8f acc[2][2];
#pragma unroll
    for (int mt = 0; mt < 2; ++mt)
#pragma unroll
      for (int nt = 0; nt < 2; ++nt) acc[mt][nt] = VZ8;
    for (int k0 = 0; k0 < DD; k0 += 32) {
      v16bf a0 = load_a16x32(sb, DD, 0, k0);
      v16bf a1 = load_a16x32(sb, DD, 16, k0);
#pragma unroll
      for (int nt = 0; nt < 2; ++nt) {
        v16bf bf = load_bp(Wb, (k0 >> 5) * 8 + wave * 2 + nt);
        acc[0][nt] = wmma_bf16(a0, bf, acc[0][nt]);
        acc[1][nt] = wmma_bf16(a1, bf, acc[1][nt]);
      }
    }
    __syncthreads(); // all reads of sb done before in-place update

#pragma unroll
    for (int nt = 0; nt < 2; ++nt) {
      const int col = wave * 32 + nt * 16 + cn;
      const float ga = sG[blk * DD + col] * BN_INV, be = sBt[blk * DD + col], ci = cival[col];
      float pmax = 0.f;
      float v0[8], v1[8];
#pragma unroll
      for (int v = 0; v < 8; ++v) {
        v0[v] = fmaxf(acc[0][nt][v] * ga + be, 0.f) * ci;
        v1[v] = fmaxf(acc[1][nt][v] * ga + be, 0.f) * ci;
        pmax = fmaxf(pmax, fmaxf(v0[v], v1[v]));
      }
      pmax = fmaxf(pmax, __shfl_xor(pmax, 16));
#pragma unroll
      for (int v = 0; v < 8; ++v) {
        const int r0 = mo + v, r1 = 16 + mo + v;
        float n0 = 0.5f * s32[r0 * DD + col] + 0.5f * v0[v];
        float n1 = 0.5f * s32[r1 * DD + col] + 0.5f * v1[v];
        s32[r0 * DD + col] = n0; sb[r0 * DD + col] = (__bf16)n0;
        s32[r1 * DD + col] = n1; sb[r1 * DD + col] = (__bf16)n1;
      }
      if ((lane & 16) == 0) aggc[col] = pmax;
    }
    __syncthreads();
    if (tid < DD) cvec[tid] = 0.5f * cvec[tid] + 0.5f * aggc[tid];
    __syncthreads();
  }
  if (tid < DD) outp[b * 384 + 256 + tid] = cvec[tid]; // mcg_emb slice
}

// =====================================================================
// Kernel 5: gather per-type LSTM hidden states + H-matrices (ones, A=1)
// =====================================================================
__global__ void gather_kernel(const int* __restrict__ obj_types,
                              const float* __restrict__ lstm_all,
                              const float* __restrict__ dlstm_all,
                              float* __restrict__ agent, float* __restrict__ Hout) {
  const int i = blockIdx.x * blockDim.x + threadIdx.x;
  if (i < BB * LHID) {
    const int b = i >> 7, d = i & 127;
    const int t = obj_types[b];
    agent[b * 384 + d]       = lstm_all[(BB * t + b) * LHID + d];
    agent[b * 384 + 128 + d] = dlstm_all[(BB * t + b) * LHID + d];
  }
  if (i < BB * 2) Hout[i] = 1.f;
}

// =====================================================================
extern "C" void kernel_launch(void* const* d_in, const int* in_sizes, int n_in,
                              void* d_out, int out_size, void* d_ws, size_t ws_size,
                              hipStream_t stream) {
  const float* map_polylines   = (const float*)d_in[0];
  const unsigned char* mp_mask = (const unsigned char*)d_in[1];
  const int* obj_types         = (const int*)d_in[2];
  // d_in[3] scatter_numbers (ones), d_in[4] scatter_idx (arange): identity -> unused
  const float* lstm_data       = (const float*)d_in[5];
  const float* lstm_data_diff  = (const float*)d_in[6];
  const float* mcg_data        = (const float*)d_in[7];
  const float* pe_pre_W  = (const float*)d_in[8];
  const float* pe_pre_g  = (const float*)d_in[9];
  const float* pe_pre_b  = (const float*)d_in[10];
  const float* pe_mlp_W1 = (const float*)d_in[11];
  const float* pe_mlp_g1 = (const float*)d_in[12];
  const float* pe_mlp_b1 = (const float*)d_in[13];
  const float* pe_mlp_W2 = (const float*)d_in[14];
  const float* pe_mlp_g2 = (const float*)d_in[15];
  const float* pe_mlp_b2 = (const float*)d_in[16];
  const float* pe_out_W  = (const float*)d_in[17];
  const float* pe_out_b  = (const float*)d_in[18];
  const float* lstm_Wx   = (const float*)d_in[19];
  const float* lstm_Wh   = (const float*)d_in[20];
  const float* lstm_b    = (const float*)d_in[21];
  const float* dlstm_Wx  = (const float*)d_in[22];
  const float* dlstm_Wh  = (const float*)d_in[23];
  const float* dlstm_b   = (const float*)d_in[24];
  const float* mcg_s_W   = (const float*)d_in[25];
  const float* mcg_s_g   = (const float*)d_in[26];
  const float* mcg_s_b   = (const float*)d_in[27];
  const float* mcg_c_W   = (const float*)d_in[28];
  const float* mcg_c_g   = (const float*)d_in[29];
  const float* mcg_c_b   = (const float*)d_in[30];

  float* out   = (float*)d_out;
  float* agent = out;                           // (2048, 384)
  float* polyf = out + (size_t)BB * 384;        // (24576, 256)
  float* Hout  = polyf + (size_t)NPOLY * OUTF;  // (2048, 2)

  float* pooled2   = (float*)d_ws;                  // 24576*64
  float* validf    = pooled2 + (size_t)NPOLY * 64;  // 24576
  float* lstm_all  = validf + NPOLY;                // 3*2048*128
  float* dlstm_all = lstm_all + (size_t)3 * BB * LHID;

  pe_encode_kernel<<<NPOLY, 128, 0, stream>>>(
      map_polylines, mp_mask, pe_pre_W, pe_pre_g, pe_pre_b,
      pe_mlp_W1, pe_mlp_g1, pe_mlp_b1, pe_mlp_W2, pe_mlp_g2, pe_mlp_b2,
      pooled2, validf);

  pe_out_kernel<<<NPOLY / 16, 128, 0, stream>>>(pooled2, validf, pe_out_W, pe_out_b, polyf);

  // ~209KB dynamic LDS per workgroup (CDNA5 WGP: 320KB)
  const size_t lstm_smem = 131072 + 32768 + 5120 + 4096 + 8192 + 32768;
  lstm_kernel<<<dim3(BB / 16, 3), 128, lstm_smem, stream>>>(lstm_data, lstm_Wx, lstm_Wh, lstm_b, lstm_all);
  lstm_kernel<<<dim3(BB / 16, 3), 128, lstm_smem, stream>>>(lstm_data_diff, dlstm_Wx, dlstm_Wh, dlstm_b, dlstm_all);

  mcg_kernel<<<BB, 128, 0, stream>>>(mcg_data, mcg_s_W, mcg_s_g, mcg_s_b,
                                     mcg_c_W, mcg_c_g, mcg_c_b, agent);

  gather_kernel<<<(BB * LHID + 255) / 256, 256, 0, stream>>>(obj_types, lstm_all, dlstm_all, agent, Hout);
}
